// GeoAttention_6871947673654
// MI455X (gfx1250) — matrix operand order repaired and checked by reading, not compiled, not run on hardware
//
#include <hip/hip_runtime.h>

#define DM    512
#define HEADS 8
#define DK    64
#define NSEQ  2048

typedef __attribute__((ext_vector_type(16))) __bf16 v16bf;
typedef __attribute__((ext_vector_type(8)))  __bf16 v8bf;
typedef __attribute__((ext_vector_type(8)))  float  v8f;

__device__ __forceinline__ v8f wmma_bf16(v16bf a, v16bf b, v8f c) {
  // D = A*B + C, 16x16x32 bf16 -> f32
  return __builtin_amdgcn_wmma_f32_16x16x32_bf16(false, a, false, b, (short)0, c,
                                                 false, false);
}

__device__ __forceinline__ v16bf cat8(v8bf lo, v8bf hi) {
  v16bf r;
#pragma unroll
  for (int j = 0; j < 8; ++j) { r[j] = lo[j]; r[j + 8] = hi[j]; }
  return r;
}

// Low 32 bits of a generic pointer into __shared__ = wave-relative LDS byte
// address (flat LDS aperture keeps the offset in addr[31:0], ISA 10.2).
__device__ __forceinline__ unsigned lds_off(const void* p) {
  return (unsigned)(size_t)p;
}

// One 32-key chunk of K ([32][64] bf16, 4KB) and V^T ([64][32] bf16, 4KB)
// staged into LDS with 16 async DMA ops (512B each), tracked by ASYNCcnt.
__device__ __forceinline__ void async_stage_chunk(
    const __bf16* __restrict__ Krow, const __bf16* __restrict__ VT, int kb,
    __bf16* ldsK, __bf16* ldsV, int lane) {
  {
    // K rows are 128B: 8 lanes per key row, 4 keys per instruction
    const int key = lane >> 3;
    const int col = (lane & 7) * 8;
#pragma unroll
    for (int i = 0; i < 8; ++i) {
      const __bf16* g = Krow + (size_t)(kb + i * 4 + key) * DK + col;
      unsigned l = lds_off(ldsK + (i * 4 + key) * DK + col);
      asm volatile("global_load_async_to_lds_b128 %0, %1, off"
                   :: "v"(l), "v"(g) : "memory");
    }
  }
  {
    // V^T chunk rows are 64B: 4 lanes per dk row, 8 rows per instruction
    const int row = lane >> 2;
    const int col = (lane & 3) * 8;
#pragma unroll
    for (int i = 0; i < 8; ++i) {
      const __bf16* g = VT + (size_t)(i * 8 + row) * NSEQ + kb + col;
      unsigned l = lds_off(ldsV + (i * 8 + row) * 32 + col);
      asm volatile("global_load_async_to_lds_b128 %0, %1, off"
                   :: "v"(l), "v"(g) : "memory");
    }
  }
}

// ---------------------------------------------------------------------------
// Kernel 1: build bf16 transposed projection weights WT[c][d] = proj[h][d][k]
// (c = h*64+k) and bf16 copy of w_out (row-major [i][j], already B-friendly).
// ---------------------------------------------------------------------------
__global__ void geo_prep_w(const float* __restrict__ qp, const float* __restrict__ kp,
                           const float* __restrict__ vp, const float* __restrict__ wo,
                           __bf16* __restrict__ WqT, __bf16* __restrict__ WkT,
                           __bf16* __restrict__ WvT, __bf16* __restrict__ Wob) {
  int idx = blockIdx.x * blockDim.x + threadIdx.x;
  if (idx >= DM * DM) return;
  int c = idx >> 9;            // 0..511 output channel
  int d = idx & (DM - 1);      // 0..511 input channel
  int h = c >> 6, kk = c & 63;
  size_t src = ((size_t)h * DM + d) * DK + kk;
  WqT[idx] = (__bf16)qp[src];
  WkT[idx] = (__bf16)kp[src];
  WvT[idx] = (__bf16)vp[src];
  Wob[idx] = (__bf16)wo[idx];
}

// ---------------------------------------------------------------------------
// Kernel 2: QKV projection. One wave computes a 16x64 output tile of the
// [B*N, 512] x [512, 512] GEMM (heads concatenated, c = h*64+k).
// Q,K stored row-major [b][h][n][dk]; V stored transposed [b][h][dk][n].
// ---------------------------------------------------------------------------
__global__ __launch_bounds__(32) void geo_qkv_proj(
    const float* __restrict__ q, const float* __restrict__ k, const float* __restrict__ v,
    const __bf16* __restrict__ WqT, const __bf16* __restrict__ WkT,
    const __bf16* __restrict__ WvT,
    const float* __restrict__ qbias, const float* __restrict__ kbias,
    const float* __restrict__ vbias,
    __bf16* __restrict__ Qb, __bf16* __restrict__ Kb, __bf16* __restrict__ VbT) {
  const int lane = threadIdx.x;
  const int lrow = lane & 15, lhi = lane >> 4;
  const int which = blockIdx.z;
  const float*  X    = (which == 0) ? q : (which == 1) ? k : v;
  const __bf16* WT   = (which == 0) ? WqT : (which == 1) ? WkT : WvT;
  const float*  bias = (which == 0) ? qbias : (which == 1) ? kbias : vbias;

  const int mBase = blockIdx.x * 16;   // row tile in [0, B*N)
  const int cBase = blockIdx.y * 64;   // output-channel tile

  v8f acc[4];
#pragma unroll
  for (int t = 0; t < 4; ++t)
#pragma unroll
    for (int r = 0; r < 8; ++r) acc[t][r] = 0.0f;

  const float* arow = X + (size_t)(mBase + lrow) * DM;
  for (int kc = 0; kc < DM; kc += 32) {
    // A fragment: lane holds row m=lrow, K elems kc+lhi*16+j and kc+lhi*16+8+j
    v16bf a;
#pragma unroll
    for (int j = 0; j < 8; ++j) {
      a[j]     = (__bf16)arow[kc + lhi * 16 + j];
      a[j + 8] = (__bf16)arow[kc + lhi * 16 + 8 + j];
    }
#pragma unroll
    for (int t = 0; t < 4; ++t) {
      // B fragment: lane n = lrow, contiguous reduction elems from WT[c][d]
      const __bf16* brow = WT + (size_t)(cBase + t * 16 + lrow) * DM + kc + lhi * 16;
      v16bf bf = *(const v16bf*)brow;
      acc[t] = wmma_bf16(a, bf, acc[t]);
    }
  }

  const int gm0 = mBase + lhi * 8;        // first of 8 rows for this lane half
  const int bb  = gm0 >> 11;              // batch (N = 2048)
  const int n0  = gm0 & (NSEQ - 1);
#pragma unroll
  for (int t = 0; t < 4; ++t) {
    const int gc = cBase + t * 16 + lrow;
    const int h = gc >> 6, dk = gc & 63;
    const float bv = bias[gc];
    if (which == 2) {
      // V transposed: consecutive r -> consecutive n -> pack one b128 store
      v8bf pk;
#pragma unroll
      for (int r = 0; r < 8; ++r) pk[r] = (__bf16)(acc[t][r] + bv);
      *(v8bf*)(VbT + (((size_t)bb * HEADS + h) * DK + dk) * NSEQ + n0) = pk;
    } else {
      __bf16* O = (which == 0) ? Qb : Kb;
#pragma unroll
      for (int r = 0; r < 8; ++r)
        O[(((size_t)bb * HEADS + h) * NSEQ + n0 + r) * DK + dk] = (__bf16)(acc[t][r] + bv);
    }
  }
}

// ---------------------------------------------------------------------------
// Kernel 3: fused flash attention with geometric RBF bias.
// One wave: 16 query rows, sweeps 2048 keys in 32-key chunks that are
// double-buffered in LDS via global_load_async_to_lds_b128 (ASYNCcnt).
// ---------------------------------------------------------------------------
__global__ __launch_bounds__(32) void geo_attn(
    const __bf16* __restrict__ Qb, const __bf16* __restrict__ Kb,
    const __bf16* __restrict__ VbT,
    const float* __restrict__ coords, const unsigned char* __restrict__ maskp,
    const float* __restrict__ spread_w, const float* __restrict__ beta_w,
    __bf16* __restrict__ Xb) {
  __shared__ __align__(16) __bf16 ldsK[2][32 * DK];   // [key][dk]   4KB x2
  __shared__ __align__(16) __bf16 ldsV[2][DK * 32];   // [dk][key]   4KB x2
  __shared__ __align__(32) __bf16 pLds[16 * 32];      // P tile D->A staging

  const int lane = threadIdx.x;
  const int lrow = lane & 15, lhi = lane >> 4;
  const int qt = blockIdx.x, h = blockIdx.y, b = blockIdx.z;
  const size_t bh = (size_t)b * HEADS + h;

  const __bf16* Qrow = Qb + (bh * NSEQ + (size_t)qt * 16) * DK;
  const __bf16* Krow = Kb + bh * NSEQ * DK;
  const __bf16* VT   = VbT + bh * DK * NSEQ;
  const float*  C    = coords + (size_t)b * NSEQ * 3;
  const unsigned char* M = maskp + (size_t)b * NSEQ;

  const float spread = 2.0f + __expf(spread_w[h]);
  const float beta   = __expf(beta_w[h]);
  const float inv2s2 = 1.0f / (2.0f * spread * spread);

  // kick off the first K/V chunk DMA before doing register setup
  async_stage_chunk(Krow, VT, 0, &ldsK[0][0], &ldsV[0][0], lane);

  // Q A-fragments for dk-chunks [0,32) and [32,64)
  v16bf qa[2];
#pragma unroll
  for (int kc = 0; kc < 2; ++kc) {
    const __bf16* ar = Qrow + (size_t)lrow * DK + kc * 32;
    v8bf lo = *(const v8bf*)(ar + lhi * 16);
    v8bf hi = *(const v8bf*)(ar + lhi * 16 + 8);
    qa[kc] = cat8(lo, hi);
  }

  // per-lane query coords: C/D-layout rows m = qt*16 + lhi*8 + r
  float qx[8], qy[8], qz[8];
#pragma unroll
  for (int r = 0; r < 8; ++r) {
    int qm = qt * 16 + lhi * 8 + r;
    qx[r] = C[qm * 3 + 0]; qy[r] = C[qm * 3 + 1]; qz[r] = C[qm * 3 + 2];
  }

  float rmax[8], rsum[8];
  v8f o[4];
#pragma unroll
  for (int r = 0; r < 8; ++r) { rmax[r] = -1e30f; rsum[r] = 0.0f; }
#pragma unroll
  for (int t = 0; t < 4; ++t)
#pragma unroll
    for (int r = 0; r < 8; ++r) o[t][r] = 0.0f;

  for (int kb = 0; kb < NSEQ; kb += 32) {
    const int cur = (kb >> 5) & 1;
    if (kb + 32 < NSEQ) {
      // issue next chunk into the other buffer, then wait for current chunk
      async_stage_chunk(Krow, VT, kb + 32, &ldsK[cur ^ 1][0], &ldsV[cur ^ 1][0], lane);
      asm volatile("s_wait_asynccnt 0x10" ::: "memory");  // <=16: cur chunk done
    } else {
      asm volatile("s_wait_asynccnt 0x0" ::: "memory");
    }
    const __bf16* Kc = &ldsK[cur][0];
    const __bf16* Vc = &ldsV[cur][0];

    // S = Q K^T : two 16x16 f32 tiles over this 32-key chunk (from LDS)
    v8f s[2];
#pragma unroll
    for (int st = 0; st < 2; ++st) {
#pragma unroll
      for (int r = 0; r < 8; ++r) s[st][r] = 0.0f;
#pragma unroll
      for (int kc = 0; kc < 2; ++kc) {
        const __bf16* kr = Kc + (st * 16 + lrow) * DK + kc * 32 + lhi * 16;
        v16bf kf = *(const v16bf*)kr;
        s[st] = wmma_bf16(qa[kc], kf, s[st]);
      }
    }

    // scale + geometric RBF bias + mask; track chunk row-max
    float chmax[8];
#pragma unroll
    for (int r = 0; r < 8; ++r) chmax[r] = -1e30f;
#pragma unroll
    for (int st = 0; st < 2; ++st) {
      const int kn = kb + st * 16 + lrow;
      const float kx = C[kn * 3 + 0], ky = C[kn * 3 + 1], kz = C[kn * 3 + 2];
      const bool mk = (M[kn] != 0);
#pragma unroll
      for (int r = 0; r < 8; ++r) {
        float sv = s[st][r] * 0.125f;  // 1/sqrt(64)
        float dx = qx[r] - kx, dy = qy[r] - ky, dz = qz[r] - kz;
        float d2 = dx * dx + dy * dy + dz * dz;
        float R  = __expf(-d2 * inv2s2);
        float rn = 2.0f * (R - 0.001f) * (1.0f / 0.999f) - 1.0f;
        sv = sv + fabsf(sv) * beta * rn;
        sv = mk ? -1e30f : sv;
        s[st][r] = sv;
        chmax[r] = fmaxf(chmax[r], sv);
      }
    }

    // online softmax: row lives in one element across a 16-lane group
#pragma unroll
    for (int r = 0; r < 8; ++r) {
#pragma unroll
      for (int w = 1; w < 16; w <<= 1)
        chmax[r] = fmaxf(chmax[r], __shfl_xor(chmax[r], w, 32));
      float nm   = fmaxf(rmax[r], chmax[r]);
      float corr = __expf(rmax[r] - nm);  // uniform across the row's lanes
      rmax[r] = nm;
      rsum[r] *= corr;
#pragma unroll
      for (int t = 0; t < 4; ++t) o[t][r] *= corr;
    }

    // P = exp(SR - rmax); per-lane partial row sums; stage to LDS [m][key]
#pragma unroll
    for (int st = 0; st < 2; ++st) {
#pragma unroll
      for (int r = 0; r < 8; ++r) {
        float p = __expf(s[st][r] - rmax[r]);
        rsum[r] += p;
        pLds[(lhi * 8 + r) * 32 + st * 16 + lrow] = (__bf16)p;
      }
    }
    __syncthreads();

    // read back as A-fragment (transpose D-layout -> A-layout via LDS)
    v16bf pa;
    {
      const __bf16* pr = pLds + lrow * 32;
      v8bf lo = *(const v8bf*)(pr + lhi * 16);
      v8bf hi = *(const v8bf*)(pr + lhi * 16 + 8);
      pa = cat8(lo, hi);
    }
    __syncthreads();

    // O += P @ V  (V^T rows in LDS give contiguous B-fragment reads)
#pragma unroll
    for (int t = 0; t < 4; ++t) {
      const __bf16* vr = Vc + (t * 16 + lrow) * 32 + lhi * 16;
      v16bf vf = *(const v16bf*)vr;
      o[t] = wmma_bf16(pa, vf, o[t]);
    }
  }

  // finalize: full row sums (butterfly all-reduce), divide, store with the
  // reference's channel interleave c = dk*HEADS + h
  float inv[8];
#pragma unroll
  for (int r = 0; r < 8; ++r) {
#pragma unroll
    for (int w = 1; w < 16; w <<= 1) rsum[r] += __shfl_xor(rsum[r], w, 32);
    inv[r] = (rsum[r] > 0.0f) ? 1.0f / rsum[r] : 0.0f;
  }
#pragma unroll
  for (int t = 0; t < 4; ++t) {
    const int dk = t * 16 + lrow;
#pragma unroll
    for (int r = 0; r < 8; ++r) {
      const size_t row = (size_t)b * NSEQ + (size_t)qt * 16 + lhi * 8 + r;
      Xb[row * DM + dk * HEADS + h] = (__bf16)(o[t][r] * inv[r]);
    }
  }
}

// ---------------------------------------------------------------------------
// Kernel 4: output projection out[m][i] = sum_j X[m][j] * w_out[i][j].
// One wave computes a 16x64 tile; writes fp32 result.
// ---------------------------------------------------------------------------
__global__ __launch_bounds__(32) void geo_out_gemm(
    const __bf16* __restrict__ Xb, const __bf16* __restrict__ Wo,
    float* __restrict__ out) {
  const int lane = threadIdx.x;
  const int lrow = lane & 15, lhi = lane >> 4;
  const int mBase = blockIdx.x * 16;
  const int iBase = blockIdx.y * 64;

  v8f acc[4];
#pragma unroll
  for (int t = 0; t < 4; ++t)
#pragma unroll
    for (int r = 0; r < 8; ++r) acc[t][r] = 0.0f;

  const __bf16* ar = Xb + (size_t)(mBase + lrow) * DM;
  for (int jc = 0; jc < DM; jc += 32) {
    v8bf lo = *(const v8bf*)(ar + jc + lhi * 16);
    v8bf hi = *(const v8bf*)(ar + jc + lhi * 16 + 8);
    v16bf a = cat8(lo, hi);
#pragma unroll
    for (int t = 0; t < 4; ++t) {
      const __bf16* br = Wo + (size_t)(iBase + t * 16 + lrow) * DM + jc + lhi * 16;
      v16bf bf = *(const v16bf*)br;
      acc[t] = wmma_bf16(a, bf, acc[t]);
    }
  }
#pragma unroll
  for (int t = 0; t < 4; ++t) {
    const int gi = iBase + t * 16 + lrow;
#pragma unroll
    for (int r = 0; r < 8; ++r)
      out[(size_t)(mBase + lhi * 8 + r) * DM + gi] = acc[t][r];
  }
}

// ---------------------------------------------------------------------------
extern "C" void kernel_launch(void* const* d_in, const int* in_sizes, int n_in,
                              void* d_out, int out_size, void* d_ws, size_t ws_size,
                              hipStream_t stream) {
  const float* q        = (const float*)d_in[0];
  const float* k        = (const float*)d_in[1];
  const float* v        = (const float*)d_in[2];
  const float* coords   = (const float*)d_in[3];
  const unsigned char* maskp = (const unsigned char*)d_in[4];
  const float* spread_w = (const float*)d_in[5];
  const float* beta_w   = (const float*)d_in[6];
  const float* q_proj   = (const float*)d_in[7];
  const float* k_proj   = (const float*)d_in[8];
  const float* v_proj   = (const float*)d_in[9];
  const float* q_bias   = (const float*)d_in[10];
  const float* k_bias   = (const float*)d_in[11];
  const float* v_bias   = (const float*)d_in[12];
  const float* w_out    = (const float*)d_in[13];
  float* out = (float*)d_out;

  const int BN = in_sizes[0] / DM;   // B * N
  const int B  = BN / NSEQ;

  char* w = (char*)d_ws;
  auto alloc = [&](size_t bytes) -> char* {
    char* p = w;
    w += (bytes + 255) & ~(size_t)255;
    return p;
  };
  __bf16* WqT = (__bf16*)alloc((size_t)DM * DM * sizeof(__bf16));
  __bf16* WkT = (__bf16*)alloc((size_t)DM * DM * sizeof(__bf16));
  __bf16* WvT = (__bf16*)alloc((size_t)DM * DM * sizeof(__bf16));
  __bf16* Wob = (__bf16*)alloc((size_t)DM * DM * sizeof(__bf16));
  __bf16* Qb  = (__bf16*)alloc((size_t)B * HEADS * NSEQ * DK * sizeof(__bf16));
  __bf16* Kb  = (__bf16*)alloc((size_t)B * HEADS * NSEQ * DK * sizeof(__bf16));
  __bf16* VbT = (__bf16*)alloc((size_t)B * HEADS * NSEQ * DK * sizeof(__bf16));
  __bf16* Xb  = (__bf16*)alloc((size_t)B * NSEQ * DM * sizeof(__bf16));

  geo_prep_w<<<(DM * DM + 255) / 256, 256, 0, stream>>>(
      q_proj, k_proj, v_proj, w_out, WqT, WkT, WvT, Wob);

  geo_qkv_proj<<<dim3(BN / 16, DM / 64, 3), 32, 0, stream>>>(
      q, k, v, WqT, WkT, WvT, q_bias, k_bias, v_bias, Qb, Kb, VbT);

  geo_attn<<<dim3(NSEQ / 16, HEADS, B), 32, 0, stream>>>(
      Qb, Kb, VbT, coords, maskp, spread_w, beta_w, Xb);

  geo_out_gemm<<<dim3(BN / 16, DM / 64), 32, 0, stream>>>(Xb, Wob, out);
}
